// SelfAttention_80642305950583
// MI455X (gfx1250) — matrix operand order repaired and checked by reading, not compiled, not run on hardware
//
#include <hip/hip_runtime.h>
#include <stdint.h>

typedef __attribute__((ext_vector_type(16))) __bf16 v16bf;
typedef __attribute__((ext_vector_type(8)))  float  v8f;
typedef unsigned int u32x4 __attribute__((ext_vector_type(4)));
typedef int          i32x8 __attribute__((ext_vector_type(8)));
typedef int          i32x4 __attribute__((ext_vector_type(4)));

#define B_   8
#define C_   256
#define CP_  32
#define N_   4096
#define NT_  64     // query rows per workgroup
#define MT_  64     // key columns per iteration
#define VPAD 8      // bf16 pad per LDS V row -> 72-ushort (144B) stride, conflict-free b128 reads

__device__ __forceinline__ unsigned short f2bf(float f) {
  unsigned int u = __float_as_uint(f);
  u += 0x7FFFu + ((u >> 16) & 1u);          // round-to-nearest-even
  return (unsigned short)(u >> 16);
}

union BF16Frag { v16bf v; uint4 q[2]; unsigned short h[16]; };
union F32Acc  { v8f  v; float f[8]; };

#if __has_builtin(__builtin_amdgcn_tensor_load_to_lds)
#define USE_TDM 1
#else
#define USE_TDM 0
#endif

// ---------------------------------------------------------------------------
// Stage v[b, 0:256, m0:m0+64] (bf16, row stride N_) into LDS buffer at byte
// offset lds_off, with 16B padding after each 128B row (=> 72-ushort rows).
// TDM path: single tensor_load_to_lds per tile (TENSORcnt, wave 0 issues).
// Fallback: global_load_async_to_lds_b128 (ASYNCcnt, all 128 threads issue).
// ---------------------------------------------------------------------------
__device__ __forceinline__ void stage_vtile(const unsigned short* vp_base, int m0,
                                            unsigned lds_off, int tid) {
#if USE_TDM
  if ((tid >> 5) == 0) {
    unsigned long long ga = (unsigned long long)(uintptr_t)vp_base
                          + (unsigned long long)m0 * 2ull;
    u32x4 g0;
    g0[0] = 1u;                                   // count=1, user descriptor
    g0[1] = lds_off;                              // lds_addr
    g0[2] = (unsigned)(ga & 0xFFFFFFFFu);         // global_addr[31:0]
    g0[3] = (unsigned)((ga >> 32) & 0x1FFFFFFu)   // global_addr[56:32]
          | (2u << 30);                           // type = 2 (image)
    i32x8 g1;
    g1[0] = (int)((1u << 16)                      // data_size = 1 -> 2 bytes
                | (1u << 20)                      // pad_enable
                | (4u << 22)                      // pad_interval: 32 DWORDs (128B row)
                | (3u << 25));                    // pad_amount: 4 DWORDs (16B)
    g1[1] = (int)((unsigned)(N_ & 0xFFFF) << 16); // tensor_dim0[15:0] @bit48
    g1[2] = (int)(((unsigned)N_ >> 16)            // tensor_dim0[31:16]
                | ((unsigned)(C_ & 0xFFFF) << 16)); // tensor_dim1[15:0] @bit80
    g1[3] = (int)(((unsigned)C_ >> 16)            // tensor_dim1[31:16]
                | (64u << 16));                   // tile_dim0 = 64 @bit112
    g1[4] = (int)(256u);                          // tile_dim1 = 256; tile_dim2 = 0
    g1[5] = (int)(N_);                            // tensor_dim0_stride[31:0]
    g1[6] = 0;                                    // stride hi, tensor_dim1_stride lo
    g1[7] = 0;
    i32x4 z = {0, 0, 0, 0};
#if __clang_major__ >= 23
    i32x8 z8 = {0, 0, 0, 0, 0, 0, 0, 0};
    __builtin_amdgcn_tensor_load_to_lds(g0, g1, z, z, z8, 0);
#else
    __builtin_amdgcn_tensor_load_to_lds(g0, g1, z, z, 0);
#endif
  }
#else
  // 2048 x 16B transfers / 128 threads = 16 per thread (16 async ops per wave)
#pragma unroll
  for (int i = 0; i < 16; ++i) {
    const int flat = i * 128 + tid;
    const int c = flat >> 3;           // channel row
    const int seg = flat & 7;          // 8 x 16B segments per 128B row
    unsigned loff = lds_off + (unsigned)(c * (MT_ + VPAD) + seg * 8) * 2u;
    unsigned goff = (unsigned)(c * N_ + m0 + seg * 8) * 2u;
    asm volatile("global_load_async_to_lds_b128 %0, %1, %2"
                 :: "v"(loff), "v"(goff), "s"(vp_base) : "memory");
  }
#endif
}

__device__ __forceinline__ void vtile_wait_one() {
#if USE_TDM
  __builtin_amdgcn_s_wait_tensorcnt((short)1);
#else
  asm volatile("s_wait_asynccnt 16" ::: "memory");
#endif
}
__device__ __forceinline__ void vtile_wait_zero() {
#if USE_TDM
  __builtin_amdgcn_s_wait_tensorcnt((short)0);
#else
  asm volatile("s_wait_asynccnt 0" ::: "memory");
#endif
}

// ---------------------------------------------------------------------------
// Kernel 1: convert fp32 weights to bf16 once.
// ---------------------------------------------------------------------------
__global__ void convert_weights(const float* __restrict__ wq,
                                const float* __restrict__ wk,
                                const float* __restrict__ wv,
                                unsigned short* __restrict__ wqb,
                                unsigned short* __restrict__ wkb,
                                unsigned short* __restrict__ wvb) {
  int i = blockIdx.x * blockDim.x + threadIdx.x;
  if (i < CP_ * C_) { wqb[i] = f2bf(wq[i]); wkb[i] = f2bf(wk[i]); }
  if (i < C_ * C_)  { wvb[i] = f2bf(wv[i]); }
}

// ---------------------------------------------------------------------------
// Kernel 2: QKV projection with WMMA.
//   q  -> (B, N, CP) bf16, kT -> (B, N, CP) bf16, v -> (B, C, N) bf16
// ---------------------------------------------------------------------------
__global__ __launch_bounds__(128) void qkv_proj(
    const float* __restrict__ x,
    const unsigned short* __restrict__ wqb, const float* __restrict__ bq,
    const unsigned short* __restrict__ wkb, const float* __restrict__ bk,
    const unsigned short* __restrict__ wvb, const float* __restrict__ bv,
    unsigned short* __restrict__ qb,
    unsigned short* __restrict__ ktb,
    unsigned short* __restrict__ vbm) {
  __shared__ unsigned short a_lds[NT_][C_ + 8];

  const int b   = blockIdx.y;
  const int n0  = blockIdx.x * NT_;
  const int tid = threadIdx.x;

  // Stage x[b, :, n0:n0+64] transposed into LDS as bf16: a_lds[n][c].
  for (int cbase = 0; cbase < C_; cbase += 8) {
    const int c = cbase + (tid >> 4);
    const int j = (tid & 15) * 4;
    const float4 xv = *(const float4*)&x[((size_t)b * C_ + c) * N_ + n0 + j];
    a_lds[j + 0][c] = f2bf(xv.x);
    a_lds[j + 1][c] = f2bf(xv.y);
    a_lds[j + 2][c] = f2bf(xv.z);
    a_lds[j + 3][c] = f2bf(xv.w);
  }
  __syncthreads();

  const int wave = tid >> 5, lane = tid & 31;
  const int hi = lane >> 4, l16 = lane & 15;
  const int kb = hi * 16;
  const int rowA = wave * 16 + l16;

  BF16Frag af[8];
#pragma unroll
  for (int kk = 0; kk < 8; ++kk) {
    af[kk].q[0] = *(const uint4*)&a_lds[rowA][kk * 32 + kb];
    af[kk].q[1] = *(const uint4*)&a_lds[rowA][kk * 32 + 8 + kb];
  }

#pragma unroll
  for (int dt = 0; dt < 2; ++dt) {
    const int d = dt * 16 + l16;
    F32Acc aq, ak;
#pragma unroll
    for (int r = 0; r < 8; ++r) { aq.f[r] = 0.f; ak.f[r] = 0.f; }
#pragma unroll
    for (int kk = 0; kk < 8; ++kk) {
      BF16Frag bfq, bfk;
      const unsigned short* wpq = &wqb[(size_t)d * C_ + kk * 32 + hi * 16];
      const unsigned short* wpk = &wkb[(size_t)d * C_ + kk * 32 + hi * 16];
      bfq.q[0] = *(const uint4*)wpq;  bfq.q[1] = *(const uint4*)(wpq + 8);
      bfk.q[0] = *(const uint4*)wpk;  bfk.q[1] = *(const uint4*)(wpk + 8);
      aq.v = __builtin_amdgcn_wmma_f32_16x16x32_bf16(false, af[kk].v, false, bfq.v,
                                                     (short)0, aq.v, false, false);
      ak.v = __builtin_amdgcn_wmma_f32_16x16x32_bf16(false, af[kk].v, false, bfk.v,
                                                     (short)0, ak.v, false, false);
    }
    const float biasq = bq[d], biask = bk[d];
#pragma unroll
    for (int r = 0; r < 8; ++r) {
      const int n = n0 + wave * 16 + r + hi * 8;
      qb [((size_t)b * N_ + n) * CP_ + d] = f2bf(aq.f[r] + biasq);
      ktb[((size_t)b * N_ + n) * CP_ + d] = f2bf(ak.f[r] + biask);
    }
  }

#pragma unroll
  for (int dt = 0; dt < 16; ++dt) {
    const int d = dt * 16 + l16;
    F32Acc av;
#pragma unroll
    for (int r = 0; r < 8; ++r) av.f[r] = 0.f;
#pragma unroll
    for (int kk = 0; kk < 8; ++kk) {
      BF16Frag bf;
      const unsigned short* wp = &wvb[(size_t)d * C_ + kk * 32 + hi * 16];
      bf.q[0] = *(const uint4*)wp;  bf.q[1] = *(const uint4*)(wp + 8);
      av.v = __builtin_amdgcn_wmma_f32_16x16x32_bf16(false, af[kk].v, false, bf.v,
                                                     (short)0, av.v, false, false);
    }
    const float bias = bv[d];
#pragma unroll
    for (int r = 0; r < 8; ++r) {
      const int n = n0 + wave * 16 + r + hi * 8;
      vbm[((size_t)b * C_ + d) * N_ + n] = f2bf(av.f[r] + bias);
    }
  }
}

// ---------------------------------------------------------------------------
// Kernel 3: flash attention with double-buffered TDM/async V staging.
// ---------------------------------------------------------------------------
__global__ __launch_bounds__(128) void attn(
    const unsigned short* __restrict__ qb,
    const unsigned short* __restrict__ ktb,
    const unsigned short* __restrict__ vbm,
    const float* __restrict__ x,
    const float* __restrict__ gamma_p,
    float* __restrict__ out) {
  __shared__ unsigned short vtile[2][C_][MT_ + VPAD];  // 2 x 36 KB, padded rows
  __shared__ unsigned short p_lds[4][16][MT_];         // per-wave P transpose buffer

  const int b = blockIdx.y, n0 = blockIdx.x * NT_;
  const int tid = threadIdx.x;
  const int wave = tid >> 5, lane = tid & 31;
  const int hi = lane >> 4, l16 = lane & 15;
  const int kb = hi * 16;

  const unsigned vt_off[2] = {
    (unsigned)(uintptr_t)&vtile[0][0][0],
    (unsigned)(uintptr_t)&vtile[1][0][0] };

  // Q A-fragment (16 rows x K=32), loaded once from global.
  BF16Frag qa;
  {
    const unsigned short* qp = &qb[((size_t)b * N_ + n0 + wave * 16 + l16) * CP_];
    qa.q[0] = *(const uint4*)(qp + kb);
    qa.q[1] = *(const uint4*)(qp + 8 + kb);
  }

  F32Acc o[16];
#pragma unroll
  for (int ct = 0; ct < 16; ++ct)
#pragma unroll
    for (int r = 0; r < 8; ++r) o[ct].f[r] = 0.f;

  float m_run[8], l_run[8];
#pragma unroll
  for (int r = 0; r < 8; ++r) { m_run[r] = -3.0e38f; l_run[r] = 0.f; }

  const unsigned short* kp_base = &ktb[(size_t)b * N_ * CP_];
  const unsigned short* vp_base = &vbm[(size_t)b * C_ * N_];

  // Prologue: start copy of tile 0.
  stage_vtile(vp_base, 0, vt_off[0], tid);

  for (int it = 0; it < N_ / MT_; ++it) {
    const int m0 = it * MT_;
    const int buf = it & 1;

    if (it + 1 < N_ / MT_) {
      stage_vtile(vp_base, m0 + MT_, vt_off[buf ^ 1], tid);   // overlap next copy
      __builtin_prefetch(&kp_base[(size_t)(m0 + MT_ + lane) * CP_], 0, 0);
      vtile_wait_one();    // current tile's copy has landed
    } else {
      vtile_wait_zero();
    }
    __syncthreads();       // all waves see the staged tile

    // ---- S = Q x K^T : 4 tiles of 16x16 (K-dim 32) ----
    F32Acc s[4];
#pragma unroll
    for (int j = 0; j < 4; ++j) {
      BF16Frag kf;
      const unsigned short* kp = &kp_base[(size_t)(m0 + j * 16 + l16) * CP_ + hi * 16];
      kf.q[0] = *(const uint4*)kp;
      kf.q[1] = *(const uint4*)(kp + 8);
      F32Acc z;
#pragma unroll
      for (int r = 0; r < 8; ++r) z.f[r] = 0.f;
      s[j].v = __builtin_amdgcn_wmma_f32_16x16x32_bf16(false, qa.v, false, kf.v,
                                                       (short)0, z.v, false, false);
    }

    // ---- Online softmax (half-wave reductions) ----
    float mnew[8], scale[8];
#pragma unroll
    for (int r = 0; r < 8; ++r) {
      float mx = s[0].f[r];
#pragma unroll
      for (int j = 1; j < 4; ++j) mx = fmaxf(mx, s[j].f[r]);
      mx = fmaxf(mx, __shfl_xor(mx, 1, 32));
      mx = fmaxf(mx, __shfl_xor(mx, 2, 32));
      mx = fmaxf(mx, __shfl_xor(mx, 4, 32));
      mx = fmaxf(mx, __shfl_xor(mx, 8, 32));
      mnew[r]  = fmaxf(m_run[r], mx);
      scale[r] = __expf(m_run[r] - mnew[r]);
      m_run[r] = mnew[r];
    }

    // ---- P = exp(S - m): write bf16 to LDS (C-layout -> A-layout), row sums ----
#pragma unroll
    for (int r = 0; r < 8; ++r) {
      float rs = 0.f;
#pragma unroll
      for (int j = 0; j < 4; ++j) {
        const float p = __expf(s[j].f[r] - mnew[r]);
        rs += p;
        p_lds[wave][r + hi * 8][j * 16 + l16] = f2bf(p);
      }
      rs += __shfl_xor(rs, 1, 32);
      rs += __shfl_xor(rs, 2, 32);
      rs += __shfl_xor(rs, 4, 32);
      rs += __shfl_xor(rs, 8, 32);
      l_run[r] = l_run[r] * scale[r] + rs;
    }

    // ---- Rescale running O ----
#pragma unroll
    for (int ct = 0; ct < 16; ++ct)
#pragma unroll
      for (int r = 0; r < 8; ++r) o[ct].f[r] *= scale[r];

    // ---- P as two A-fragments (K halves) ----
    BF16Frag pa0, pa1;
    pa0.q[0] = *(const uint4*)&p_lds[wave][l16][kb];
    pa0.q[1] = *(const uint4*)&p_lds[wave][l16][8 + kb];
    pa1.q[0] = *(const uint4*)&p_lds[wave][l16][32 + kb];
    pa1.q[1] = *(const uint4*)&p_lds[wave][l16][40 + kb];

    // ---- O += P x V^T over 16 channel tiles (V from staged LDS tile) ----
#pragma unroll
    for (int ct = 0; ct < 16; ++ct) {
      const unsigned short* vp = &vtile[buf][ct * 16 + l16][hi * 16];
      BF16Frag v0, v1;
      v0.q[0] = *(const uint4*)vp;        v0.q[1] = *(const uint4*)(vp + 8);
      v1.q[0] = *(const uint4*)(vp + 32); v1.q[1] = *(const uint4*)(vp + 40);
      o[ct].v = __builtin_amdgcn_wmma_f32_16x16x32_bf16(false, pa0.v, false, v0.v,
                                                        (short)0, o[ct].v, false, false);
      o[ct].v = __builtin_amdgcn_wmma_f32_16x16x32_bf16(false, pa1.v, false, v1.v,
                                                        (short)0, o[ct].v, false, false);
    }

    __syncthreads();  // all waves done reading vtile[buf] before tile it+2 overwrites it
  }

  // ---- Epilogue: normalize, gamma * o + x ----
  const float gamma = gamma_p[0];
#pragma unroll
  for (int r = 0; r < 8; ++r) {
    const float inv = 1.0f / l_run[r];
    const int n = n0 + wave * 16 + r + hi * 8;
#pragma unroll
    for (int ct = 0; ct < 16; ++ct) {
      const int c = ct * 16 + l16;
      const size_t idx = ((size_t)b * C_ + c) * N_ + n;
      out[idx] = gamma * (o[ct].f[r] * inv) + x[idx];
    }
  }
}

// ---------------------------------------------------------------------------
extern "C" void kernel_launch(void* const* d_in, const int* in_sizes, int n_in,
                              void* d_out, int out_size, void* d_ws, size_t ws_size,
                              hipStream_t stream) {
  const float* x     = (const float*)d_in[0];
  const float* Wq    = (const float*)d_in[1];
  const float* bq    = (const float*)d_in[2];
  const float* Wk    = (const float*)d_in[3];
  const float* bk    = (const float*)d_in[4];
  const float* Wv    = (const float*)d_in[5];
  const float* bv    = (const float*)d_in[6];
  const float* gamma = (const float*)d_in[7];
  float* out = (float*)d_out;

  char* ws = (char*)d_ws;
  unsigned short* wqb = (unsigned short*)(ws + 0);                    // 16 KB
  unsigned short* wkb = (unsigned short*)(ws + 16384);                // 16 KB
  unsigned short* wvb = (unsigned short*)(ws + 32768);                // 128 KB
  unsigned short* qb  = (unsigned short*)(ws + 163840);               // 2 MB
  unsigned short* ktb = (unsigned short*)(ws + 163840 + 2097152);     // 2 MB
  unsigned short* vbm = (unsigned short*)(ws + 163840 + 2 * 2097152); // 16 MB

  hipLaunchKernelGGL(convert_weights, dim3(256), dim3(256), 0, stream,
                     Wq, Wk, Wv, wqb, wkb, wvb);
  hipLaunchKernelGGL(qkv_proj, dim3(64, B_), dim3(128), 0, stream,
                     x, wqb, bq, wkb, bk, wvb, bv, qb, ktb, vbm);
  hipLaunchKernelGGL(attn, dim3(64, B_), dim3(128), 0, stream,
                     qb, ktb, vbm, x, gamma, out);
}
